// DKTForget_11038065951534
// MI455X (gfx1250) — matrix operand order repaired and checked by reading, not compiled, not run on hardware
//
#include <hip/hip_runtime.h>
#include <cstdint>
#include <cstddef>

// ---------------- problem constants (from reference) ----------------
#define NUMC  2048
#define BB    256
#define SS    512
#define EMB   256
#define NTOT  64
#define KIN   320      // EMB + NTOT
#define G4    1024     // 4*EMB
#define BS    (BB*SS)  // 131072

typedef __attribute__((ext_vector_type(16))) __bf16 v16bf;
typedef __attribute__((ext_vector_type(8)))  __bf16 v8bf;
typedef __attribute__((ext_vector_type(8)))  float  v8f;

__device__ __forceinline__ unsigned short f2bf(float f) {
  unsigned u = __float_as_uint(f);
  u += 0x7FFFu + ((u >> 16) & 1u);          // round-to-nearest-even
  return (unsigned short)(u >> 16);
}
__device__ __forceinline__ float bf2f(unsigned short h) {
  return __uint_as_float(((unsigned)h) << 16);
}
__device__ __forceinline__ float sigm(float x) {
  return 1.0f / (1.0f + __expf(-x));
}

// ---- WMMA operand loaders (bf16, row-major source, wave32 layouts) ----
__device__ __forceinline__ v16bf load_a_tile(const unsigned short* A, int lda,
                                             size_t row0, int k0, int lane) {
  int m  = lane & 15;
  int h8 = (lane >> 4) << 4;
  const unsigned short* p = A + (row0 + (size_t)m) * (size_t)lda + k0 + h8;
  v8bf lo = *reinterpret_cast<const v8bf*>(p);
  v8bf hi = *reinterpret_cast<const v8bf*>(p + 8);
  return __builtin_shufflevector(lo, hi, 0,1,2,3,4,5,6,7,8,9,10,11,12,13,14,15);
}
__device__ __forceinline__ v16bf load_b_tile(const unsigned short* W, int ldw,
                                             int n0, int k0, int lane) {
  int n  = lane & 15;
  int kh = (lane >> 4) << 4;
  return *reinterpret_cast<const v16bf*>(W + (size_t)(n0 + n) * (size_t)ldw + k0 + kh);
}

// ---------------- kernel 1: weight conversion + bias fuse ----------------
__global__ void prep_weights(const float* __restrict__ Wih, const float* __restrict__ Whh,
                             const float* __restrict__ Wout,
                             const float* __restrict__ bih, const float* __restrict__ bhh,
                             unsigned short* __restrict__ Wih_b,
                             unsigned short* __restrict__ Whh_b,
                             unsigned short* __restrict__ Wout_b,
                             float* __restrict__ bias_g) {
  int i = blockIdx.x * blockDim.x + threadIdx.x;
  if (i < G4 * KIN)   Wih_b[i]  = f2bf(Wih[i]);
  if (i < G4 * EMB)   Whh_b[i]  = f2bf(Whh[i]);
  if (i < NUMC * KIN) Wout_b[i] = f2bf(Wout[i]);
  if (i < G4)         bias_g[i] = bih[i] + bhh[i];
}

// ------------- kernel 2: theta_in = [xemb*Cct, onehot] (bf16) -------------
__global__ void build_theta_in(const int* __restrict__ q, const int* __restrict__ r,
                               const int* __restrict__ rg, const int* __restrict__ sg,
                               const int* __restrict__ pc,
                               const float* __restrict__ E, const float* __restrict__ Wc,
                               unsigned short* __restrict__ theta) {
  int site = blockIdx.x;                 // b*S + s
  int t    = threadIdx.x;                // 0..255
  int x    = q[site] + NUMC * r[site];
  int a    = rg[site], b = sg[site], p = pc[site];
  float xv  = E[(size_t)x * EMB + t];
  const float* wr = Wc + (size_t)t * NTOT;
  float cct = wr[a] + wr[16 + b] + wr[32 + p];
  size_t base = (size_t)site * KIN;
  theta[base + t] = f2bf(xv * cct);
  if (t < NTOT) {
    float oh = (t < 16) ? (t == a ? 1.f : 0.f)
             : (t < 32) ? (t - 16 == b ? 1.f : 0.f)
                        : (t - 32 == p ? 1.f : 0.f);
    theta[base + EMB + t] = f2bf(oh);
  }
}

// ------------- kernel 5: theta_out from h (bf16 in) -------------
__global__ void build_theta_out(const unsigned short* __restrict__ hseq,
                                const int* __restrict__ rg, const int* __restrict__ sg,
                                const int* __restrict__ pc,
                                const float* __restrict__ Wc,
                                unsigned short* __restrict__ theta) {
  int site = blockIdx.x;
  int t    = threadIdx.x;
  int a    = rg[site], b = sg[site], p = pc[site];
  float hv  = bf2f(hseq[(size_t)site * EMB + t]);
  const float* wr = Wc + (size_t)t * NTOT;
  float cct = wr[a] + wr[16 + b] + wr[32 + p];
  size_t base = (size_t)site * KIN;
  theta[base + t] = f2bf(hv * cct);
  if (t < NTOT) {
    float oh = (t < 16) ? (t == a ? 1.f : 0.f)
             : (t < 32) ? (t - 16 == b ? 1.f : 0.f)
                        : (t - 32 == p ? 1.f : 0.f);
    theta[base + EMB + t] = f2bf(oh);
  }
}

// ------------- WMMA GEMM: C[M,N] = A[M,K] @ W[N,K]^T + bias (opt sigmoid) -------------
template <int SIG>
__global__ __launch_bounds__(256)
void gemm_bf16_kernel(const unsigned short* __restrict__ A,
                      const unsigned short* __restrict__ W,
                      const float* __restrict__ bias,
                      float* __restrict__ C,
                      int N, int K, int lda, int ldw) {
  int lane = threadIdx.x & 31;
  int wv   = threadIdx.x >> 5;
  size_t m0 = (size_t)blockIdx.y * 128 + (size_t)wv * 16;
  int n0 = blockIdx.x * 64;

  v8f acc[4];
#pragma unroll
  for (int nt = 0; nt < 4; ++nt) acc[nt] = (v8f){0.f,0.f,0.f,0.f,0.f,0.f,0.f,0.f};

  for (int k0 = 0; k0 < K; k0 += 32) {
    if (k0 + 32 < K) {  // prefetch next K-slice (global_prefetch_b8)
      __builtin_prefetch((const void*)(A + (m0 + (lane & 15)) * (size_t)lda + k0 + 32), 0, 2);
      __builtin_prefetch((const void*)(W + (size_t)(n0 + (lane & 15)) * (size_t)ldw + k0 + 32), 0, 2);
    }
    v16bf a = load_a_tile(A, lda, m0, k0, lane);
#pragma unroll
    for (int nt = 0; nt < 4; ++nt) {
      v16bf b = load_b_tile(W, ldw, n0 + nt * 16, k0, lane);
      acc[nt] = __builtin_amdgcn_wmma_f32_16x16x32_bf16(
          false, a, false, b, (short)0, acc[nt], false, false);
    }
  }
  int n  = lane & 15;
  int mh = (lane >> 4) << 3;
#pragma unroll
  for (int nt = 0; nt < 4; ++nt) {
    int col = n0 + nt * 16 + n;
    float bv = bias[col];
#pragma unroll
    for (int j = 0; j < 8; ++j) {
      float v = acc[nt][j] + bv;
      if constexpr (SIG) v = sigm(v);
      C[(m0 + mh + j) * (size_t)N + col] = v;
    }
  }
}

// ------------- kernel 4: LSTM scan (persistent per batch-tile of 16) -------------
// 16 blocks x 512 threads (16 waves); wave w owns gate columns [w*64, w*64+64).
// W_hh: K 0..127 register-resident (breg[4][4] = 128 VGPRs/wave),
//       K 128..255 staged once in LDS for all 1024 gate rows (256 KB).
// gates bf16 in LDS (32 KB); h bf16 in LDS (8 KB); cell state c in registers.
// WMMA runs as two nt-halves (acc = 16 VGPRs). Activation uses a contiguous-8
// mapping: thread tid -> row m = tid>>5, elems e0 = (tid&31)*8, so all xg/gate/h
// traffic is b128-vectorized; i/f-gate xg prefetched before the WMMA phase.
__global__ __launch_bounds__(512, 4)
void lstm_scan(const float* __restrict__ xg,
               const unsigned short* __restrict__ Whh_b,
               unsigned short* __restrict__ hseq) {
  extern __shared__ char smem[];
  unsigned short* ls_whh   = (unsigned short*)smem;                     // 262144 B
  unsigned short* ls_gates = (unsigned short*)(smem + 262144);          //  32768 B
  unsigned short* ls_hbf   = (unsigned short*)(smem + 262144 + 32768);  //   8192 B

  int tid  = threadIdx.x;
  int lane = tid & 31;
  int wv   = tid >> 5;              // 0..15
  int b0   = blockIdx.x * 16;
  int nbase = wv * 64;

  // stage upper K-half of W_hh into LDS: ls_whh[n*128 + kk] = Whh_b[n*256 + 128 + kk]
  {
    const uint4* src = (const uint4*)Whh_b;          // 32 uint4 per 256-ushort row
    uint4* dst = (uint4*)ls_whh;                     // 16 uint4 per 128-ushort row
    for (int i = tid; i < 1024 * 16; i += 512) {
      int n = i >> 4, c = i & 15;
      dst[n * 16 + c] = src[n * 32 + 16 + c];
    }
  }
  // register-resident lower K-half (K 0..127) for this wave's N-chunk
  v16bf breg[4][4];
#pragma unroll
  for (int nt = 0; nt < 4; ++nt)
#pragma unroll
    for (int kt = 0; kt < 4; ++kt)
      breg[nt][kt] = load_b_tile(Whh_b, EMB, nbase + nt * 16, kt * 32, lane);

  // activation-phase ownership: row am, elements ae0..ae0+7
  int am  = tid >> 5;
  int ae0 = (tid & 31) * 8;
  float c_reg[8];
#pragma unroll
  for (int i = 0; i < 8; ++i) c_reg[i] = 0.f;
  for (int i = tid; i < 16 * EMB; i += 512) ls_hbf[i] = 0;
  __syncthreads();

  for (int s = 0; s < SS; ++s) {
    size_t xb = ((size_t)(b0 + am) * SS + s) * G4 + ae0;
    // prefetch i/f-gate xg (4x float4); latency hides under WMMA below
    float xi[8], xf[8];
    *(float4*)&xi[0] = *(const float4*)(xg + xb);
    *(float4*)&xi[4] = *(const float4*)(xg + xb + 4);
    *(float4*)&xf[0] = *(const float4*)(xg + xb + EMB);
    *(float4*)&xf[4] = *(const float4*)(xg + xb + EMB + 4);

    int lofs = 0;
    asm volatile("" : "+v"(lofs));   // opaque 0: keep LDS B-tile loads inside s-loop

    // gates_rec = h @ Whh^T, two nt-halves to halve accumulator pressure
    int n = lane & 15, mh = (lane >> 4) << 3;
#pragma unroll
    for (int half = 0; half < 2; ++half) {
      v8f acc[2];
      acc[0] = (v8f){0.f,0.f,0.f,0.f,0.f,0.f,0.f,0.f};
      acc[1] = (v8f){0.f,0.f,0.f,0.f,0.f,0.f,0.f,0.f};
#pragma unroll
      for (int kt = 0; kt < 4; ++kt) {          // K 0..127 from registers
        v16bf a = load_a_tile(ls_hbf, EMB, 0, kt * 32, lane);
#pragma unroll
        for (int u = 0; u < 2; ++u)
          acc[u] = __builtin_amdgcn_wmma_f32_16x16x32_bf16(
              false, a, false, breg[half * 2 + u][kt], (short)0, acc[u], false, false);
      }
#pragma unroll
      for (int kt = 0; kt < 4; ++kt) {          // K 128..255 from LDS
        v16bf a = load_a_tile(ls_hbf, EMB, 0, 128 + kt * 32, lane);
#pragma unroll
        for (int u = 0; u < 2; ++u) {
          v16bf b = load_b_tile(ls_whh + lofs, 128, nbase + (half * 2 + u) * 16, kt * 32, lane);
          acc[u] = __builtin_amdgcn_wmma_f32_16x16x32_bf16(
              false, a, false, b, (short)0, acc[u], false, false);
        }
      }
      // scatter this half's gates to LDS as bf16
#pragma unroll
      for (int u = 0; u < 2; ++u) {
        int n0 = nbase + (half * 2 + u) * 16;
#pragma unroll
        for (int j = 0; j < 8; ++j)
          ls_gates[(mh + j) * G4 + n0 + n] = f2bf(acc[u][j]);
      }
    }
    __syncthreads();

    // activation: vectorized b128 traffic, c in registers
    {
      float xgt[8], xo[8];
      *(float4*)&xgt[0] = *(const float4*)(xg + xb + 2 * EMB);
      *(float4*)&xgt[4] = *(const float4*)(xg + xb + 2 * EMB + 4);
      *(float4*)&xo[0]  = *(const float4*)(xg + xb + 3 * EMB);
      *(float4*)&xo[4]  = *(const float4*)(xg + xb + 3 * EMB + 4);

      const unsigned short* gp = ls_gates + am * G4 + ae0;
      unsigned short gi8[8], gf8[8], gg8[8], go8[8];
      *(uint4*)gi8 = *(const uint4*)(gp);
      *(uint4*)gf8 = *(const uint4*)(gp + EMB);
      *(uint4*)gg8 = *(const uint4*)(gp + 2 * EMB);
      *(uint4*)go8 = *(const uint4*)(gp + 3 * EMB);

      unsigned short hb[8];
#pragma unroll
      for (int j = 0; j < 8; ++j) {
        float gi = sigm(bf2f(gi8[j]) + xi[j]);
        float gf = sigm(bf2f(gf8[j]) + xf[j]);
        float gg = tanhf(bf2f(gg8[j]) + xgt[j]);
        float go = sigm(bf2f(go8[j]) + xo[j]);
        float c = gf * c_reg[j] + gi * gg;
        c_reg[j] = c;
        hb[j] = f2bf(go * tanhf(c));
      }
      *(uint4*)(ls_hbf + am * EMB + ae0) = *(uint4*)hb;
      *(uint4*)(hseq + ((size_t)(b0 + am) * SS + s) * EMB + ae0) = *(uint4*)hb;
    }
    __syncthreads();
  }
}

// ------------------------------- launch -------------------------------
extern "C" void kernel_launch(void* const* d_in, const int* in_sizes, int n_in,
                              void* d_out, int out_size, void* d_ws, size_t ws_size,
                              hipStream_t stream) {
  const int*   q      = (const int*)d_in[0];
  const int*   r      = (const int*)d_in[1];
  const int*   rgaps  = (const int*)d_in[2];
  const int*   sgaps  = (const int*)d_in[3];
  const int*   pcnts  = (const int*)d_in[4];
  const int*   srg    = (const int*)d_in[5];
  const int*   ssg    = (const int*)d_in[6];
  const int*   spc    = (const int*)d_in[7];
  const float* E      = (const float*)d_in[8];
  const float* Wc     = (const float*)d_in[9];
  const float* Wih    = (const float*)d_in[10];
  const float* Whh    = (const float*)d_in[11];
  const float* bih    = (const float*)d_in[12];
  const float* bhh    = (const float*)d_in[13];
  const float* Wout   = (const float*)d_in[14];
  const float* bout   = (const float*)d_in[15];
  float* y = (float*)d_out;

  char* ws = (char*)d_ws;
  size_t off = 0;
  auto alloc = [&](size_t bytes) { size_t o = off; off += (bytes + 255) & ~(size_t)255; return o; };
  unsigned short* theta_in  = (unsigned short*)(ws + alloc((size_t)BS * KIN * 2));
  float*          xg        = (float*)         (ws + alloc((size_t)BS * G4 * 4));
  unsigned short* hseq      = (unsigned short*)(ws + alloc((size_t)BS * EMB * 2));
  unsigned short* theta_out = (unsigned short*)(ws + alloc((size_t)BS * KIN * 2));
  unsigned short* Wih_b     = (unsigned short*)(ws + alloc((size_t)G4 * KIN * 2));
  unsigned short* Whh_b     = (unsigned short*)(ws + alloc((size_t)G4 * EMB * 2));
  unsigned short* Wout_b    = (unsigned short*)(ws + alloc((size_t)NUMC * KIN * 2));
  float*          bias_g    = (float*)         (ws + alloc((size_t)G4 * 4));

  // 1) weights -> bf16, fused LSTM bias
  prep_weights<<<(NUMC * KIN + 255) / 256, 256, 0, stream>>>(
      Wih, Whh, Wout, bih, bhh, Wih_b, Whh_b, Wout_b, bias_g);

  // 2) theta_in (embedding gather + c-integration)
  build_theta_in<<<BS, 256, 0, stream>>>(q, r, rgaps, sgaps, pcnts, E, Wc, theta_in);

  // 3) xg = theta_in @ Wih^T + (b_ih+b_hh)   [BS,320]x[320,1024]
  gemm_bf16_kernel<0><<<dim3(G4 / 64, BS / 128), 256, 0, stream>>>(
      theta_in, Wih_b, bias_g, xg, G4, KIN, KIN, KIN);

  // 4) LSTM scan over S (16 batch rows per block; W_hh in regs + LDS)
  lstm_scan<<<BB / 16, 512, 262144 + 32768 + 8192, stream>>>(xg, Whh_b, hseq);

  // 5) theta_out from h + shifted gaps
  build_theta_out<<<BS, 256, 0, stream>>>(hseq, srg, ssg, spc, Wc, theta_out);

  // 6) y = sigmoid(theta_out @ Wout^T + b_out)  [BS,320]x[320,2048]
  gemm_bf16_kernel<1><<<dim3(NUMC / 64, BS / 128), 256, 0, stream>>>(
      theta_out, Wout_b, bout, y, NUMC, KIN, KIN, KIN);
}
